// VariationalEncoder_43911745635209
// MI455X (gfx1250) — matrix operand order repaired and checked by reading, not compiled, not run on hardware
//
#include <hip/hip_runtime.h>
#include <hip/hip_bf16.h>
#include <cstdint>

// Problem constants (from reference)
#define NNODES 100000
#define NEDGES 1600000
#define HDIM   256
#define LDIM   64

typedef __bf16 bf16_t;
typedef __bf16 v16bf __attribute__((ext_vector_type(16)));
typedef float  v8f   __attribute__((ext_vector_type(8)));
typedef int    av4i  __attribute__((__vector_size__(16)));  // async-LDS builtin operand type

#if __has_builtin(__builtin_amdgcn_global_load_async_to_lds_b128) && \
    __has_builtin(__builtin_amdgcn_s_wait_asynccnt)
#define HAVE_ASYNC_LDS 1
#else
#define HAVE_ASYNC_LDS 0
#endif

union FragU { uint4 u[2]; v16bf v; };
union Pack8 { uint4 u; bf16_t b[8]; };

static __device__ __forceinline__ bf16_t f2b(float f) { return (bf16_t)f; }
static __device__ __forceinline__ float  b2f(bf16_t b) { return (float)b; }

// 32-byte global -> LDS tile copy.  Async (ASYNCcnt-tracked GLOBAL_LOAD_ASYNC_TO_LDS_B128)
// on gfx1250 when the builtin is available; falls back to regs + ds_store otherwise.
static __device__ __forceinline__ void stage32(bf16_t* __restrict__ dst,
                                               const bf16_t* __restrict__ src) {
#if HAVE_ASYNC_LDS
  __builtin_amdgcn_global_load_async_to_lds_b128(
      (__attribute__((address_space(1))) av4i*)(src),
      (__attribute__((address_space(3))) av4i*)(dst), 0, 0);
  __builtin_amdgcn_global_load_async_to_lds_b128(
      (__attribute__((address_space(1))) av4i*)(src),
      (__attribute__((address_space(3))) av4i*)(dst), 16, 0);
#else
  const uint4* s = (const uint4*)src;
  uint4* d = (uint4*)dst;
  d[0] = s[0];
  d[1] = s[1];
#endif
}

static __device__ __forceinline__ void async_fence() {
#if HAVE_ASYNC_LDS
  __builtin_amdgcn_s_wait_asynccnt(0);
#endif
}

// ---------------------------------------------------------------- utilities
__global__ void fill_zero_f32(float* __restrict__ p, int n) {
  int i = blockIdx.x * blockDim.x + threadIdx.x;
  if (i < n) p[i] = 0.0f;
}

// fp32 -> bf16, 8 elements per thread (L2-resident pass so GEMM staging is pure async copy)
__global__ void cvt_f32_bf16(const float* __restrict__ src, bf16_t* __restrict__ dst, int n8) {
  int i = blockIdx.x * blockDim.x + threadIdx.x;
  if (i >= n8) return;
  const float4* s = (const float4*)(src + (size_t)i * 8);
  float4 a = s[0], b = s[1];
  Pack8 p;
  p.b[0] = f2b(a.x); p.b[1] = f2b(a.y); p.b[2] = f2b(a.z); p.b[3] = f2b(a.w);
  p.b[4] = f2b(b.x); p.b[5] = f2b(b.y); p.b[6] = f2b(b.z); p.b[7] = f2b(b.w);
  ((uint4*)dst)[i] = p.u;
}

// Transpose + bf16-convert layer weights: WT[n][k], n<256, k<512
// k<256 -> w_rel[k][n], else w_root[k-256][n]
__global__ void prep_layer_w(const float* __restrict__ w_rel,
                             const float* __restrict__ w_root,
                             bf16_t* __restrict__ WT) {
  int i = blockIdx.x * blockDim.x + threadIdx.x;
  if (i >= 256 * 512) return;
  int n = i >> 9, k = i & 511;
  float v = (k < 256) ? w_rel[k * 256 + n] : w_root[(k - 256) * 256 + n];
  WT[i] = f2b(v);
}

// Heads: WTh[n][k], n<128, k<256 ; n<64 -> mu_w[k][n], else ls_w[k][n-64]
__global__ void prep_heads_w(const float* __restrict__ mu_w,
                             const float* __restrict__ ls_w,
                             const float* __restrict__ mu_b,
                             const float* __restrict__ ls_b,
                             bf16_t* __restrict__ WTh,
                             float* __restrict__ biash) {
  int i = blockIdx.x * blockDim.x + threadIdx.x;
  if (i < 128) biash[i] = (i < 64) ? mu_b[i] : ls_b[i - 64];
  if (i >= 128 * 256) return;
  int n = i >> 8, k = i & 255;
  float v = (n < 64) ? mu_w[k * 64 + n] : ls_w[k * 64 + (n - 64)];
  WTh[i] = f2b(v);
}

// ---------------------------------------------------------------- layer 1 (d_in = 5)
__global__ void aggregate_d5(const int* __restrict__ src, const int* __restrict__ dst,
                             const float* __restrict__ x, float* __restrict__ agg) {
  int e = blockIdx.x * blockDim.x + threadIdx.x;
  if (e >= NEDGES) return;
  int s = src[e], d = dst[e];
#pragma unroll
  for (int k = 0; k < 5; ++k)
    atomicAdd(&agg[d * 5 + k], x[s * 5 + k]);
}

__global__ void layer1_dense(const float* __restrict__ x, const float* __restrict__ agg,
                             const float* __restrict__ w_rel, const float* __restrict__ w_root,
                             const float* __restrict__ b, bf16_t* __restrict__ hout) {
  int i = blockIdx.x * blockDim.x + threadIdx.x;
  if (i >= NNODES * HDIM) return;
  int row = i >> 8, c = i & 255;
  float s = b[c];
#pragma unroll
  for (int k = 0; k < 5; ++k)
    s += agg[row * 5 + k] * w_rel[k * 256 + c] + x[row * 5 + k] * w_root[k * 256 + c];
  hout[i] = f2b(fmaxf(s, 0.0f));
}

// ---------------------------------------------------------------- edge aggregation (H=256)
// 16 threads per edge, 16 features each: L2-resident gather (bf16) + fp32 atomic scatter.
__global__ void aggregate_h(const int* __restrict__ src, const int* __restrict__ dst,
                            const bf16_t* __restrict__ h, float* __restrict__ agg) {
  int t = blockIdx.x * blockDim.x + threadIdx.x;
  if (t >= NEDGES * 16) return;
  int e  = t >> 4;
  int f0 = (t & 15) << 4;
  int s = src[e], d = dst[e];
  FragU raw;
  const uint4* hp = (const uint4*)(h + (size_t)s * HDIM + f0);
  raw.u[0] = hp[0];
  raw.u[1] = hp[1];
  float* base = agg + (size_t)d * HDIM + f0;
#pragma unroll
  for (int j = 0; j < 16; ++j)
    atomicAdd(base + j, b2f(raw.v[j]));
}

// ---------------------------------------------------------------- WMMA GEMM
// MODE 0: hidden layer.  C[M,256] = [aggbf | h] (K=512, both bf16) @ WT^T + bias, ReLU, bf16 out.
// MODE 1: heads.         C[M,128] = h (K=256) @ WTh^T + bias, fp32 split (mu|logstd) out.
// Block: 256 thr (8 waves, 4x2), tile 128x128, 16x16x32 bf16 WMMA fragments.
// Double-buffered LDS; K-tiles staged with GLOBAL_LOAD_ASYNC_TO_LDS_B128 (ASYNCcnt)
// overlapping the current tile's WMMAs.
template <int MODE>
__global__ void __launch_bounds__(256)
gemm_wmma(const bf16_t* __restrict__ Alo, const bf16_t* __restrict__ Ahi,
          const bf16_t* __restrict__ WT, const float* __restrict__ bias,
          bf16_t* __restrict__ outh, float* __restrict__ outf) {
  constexpr int K    = (MODE == 0) ? 512 : 256;
  constexpr int KT   = K / 32;
  constexpr int LDA  = 40;          // padded bf16 stride: conflict-free b128 LDS access
  constexpr int TILE = 128 * LDA;   // one tile in bf16 elements

  __shared__ __align__(16) bf16_t As[2 * TILE];
  __shared__ __align__(16) bf16_t Bs[2 * TILE];

  const int tid  = threadIdx.x;
  const int lane = tid & 31;
  const int wave = tid >> 5;
  const int wm   = wave & 3;   // 4 waves over M (32 rows each)
  const int wn   = wave >> 2;  // 2 waves over N (64 cols each)
  const int m0   = blockIdx.x * 128;
  const int n0   = blockIdx.y * 128;

  const int sr = tid >> 1;         // staging row 0..127
  const int sh = (tid & 1) << 4;   // staging col offset 0 / 16
  const int gm = m0 + sr;
  const bf16_t* Brow = WT + (size_t)(n0 + sr) * K + sh;

  v8f acc[2][4] = {};

  // stage one 128x32 K-tile (A + B) into buffer `buf`
  auto stage = [&](int kt, int buf) {
    const int k0 = kt * 32;
    bf16_t* ad = As + buf * TILE + sr * LDA + sh;
    if (gm < NNODES) {
      const bf16_t* asrc;
      if (MODE == 0)
        asrc = (k0 < 256) ? (Alo + (size_t)gm * 256 + k0 + sh)
                          : (Ahi + (size_t)gm * 256 + (k0 - 256) + sh);
      else
        asrc = Alo + (size_t)gm * 256 + k0 + sh;
      stage32(ad, asrc);
    } else {
      uint4 z; z.x = z.y = z.z = z.w = 0u;
      ((uint4*)ad)[0] = z;
      ((uint4*)ad)[1] = z;
    }
    stage32(Bs + buf * TILE + sr * LDA + sh, Brow + k0);
  };

  // prologue: fill buffer 0
  stage(0, 0);
  async_fence();
  __syncthreads();

  for (int kt = 0; kt < KT; ++kt) {
    const int cb = kt & 1;
    // prefetch next K-tile into the other buffer; overlaps with the WMMAs below
    if (kt + 1 < KT) stage(kt + 1, cb ^ 1);

    // ---- fragments from LDS (layouts per CDNA5 ISA 7.12.2) ----
    FragU fa[2], fb[4];
    const int g = lane >> 4;
    const bf16_t* Ab = As + cb * TILE;
    const bf16_t* Bb = Bs + cb * TILE;
#pragma unroll
    for (int mi = 0; mi < 2; ++mi) {
      const bf16_t* p = Ab + (wm * 32 + mi * 16 + (lane & 15)) * LDA + g * 8;
      fa[mi].u[0] = *(const uint4*)p;         // k = g*8 + 0..7
      fa[mi].u[1] = *(const uint4*)(p + 16);  // k = 16 + g*8 + 0..7
    }
#pragma unroll
    for (int ni = 0; ni < 4; ++ni) {
      const bf16_t* p = Bb + (wn * 64 + ni * 16 + (lane & 15)) * LDA + g * 8;
      fb[ni].u[0] = *(const uint4*)p;         // k = g*8 + 0..7
      fb[ni].u[1] = *(const uint4*)(p + 16);  // k = 16 + g*8 + 0..7
    }
#pragma unroll
    for (int mi = 0; mi < 2; ++mi)
#pragma unroll
      for (int ni = 0; ni < 4; ++ni)
        acc[mi][ni] = __builtin_amdgcn_wmma_f32_16x16x32_bf16(
            false, fa[mi].v, false, fb[ni].v, (short)0, acc[mi][ni], false, false);

    // retire this iteration's async copies, then make them visible to all waves
    async_fence();
    __syncthreads();
  }

  // ---- epilogue: bias (+ReLU + bf16) or fp32 split store ----
#pragma unroll
  for (int ni = 0; ni < 4; ++ni) {
    const int gcol = n0 + wn * 64 + ni * 16 + (lane & 15);
    const float bv = bias[gcol];
#pragma unroll
    for (int mi = 0; mi < 2; ++mi) {
      const int rbase = m0 + wm * 32 + mi * 16 + ((lane >> 4) << 3);
#pragma unroll
      for (int i = 0; i < 8; ++i) {
        const int row = rbase + i;
        if (row >= NNODES) continue;
        const float v = acc[mi][ni][i] + bv;
        if (MODE == 0) {
          outh[(size_t)row * HDIM + gcol] = f2b(fmaxf(v, 0.0f));
        } else {
          if (gcol < 64)
            outf[(size_t)row * 64 + gcol] = v;                               // mu
          else
            outf[(size_t)NNODES * 64 + (size_t)row * 64 + (gcol - 64)] = v;  // logstd
        }
      }
    }
  }
}

// ---------------------------------------------------------------- host
extern "C" void kernel_launch(void* const* d_in, const int* in_sizes, int n_in,
                              void* d_out, int out_size, void* d_ws, size_t ws_size,
                              hipStream_t stream) {
  (void)in_sizes; (void)n_in; (void)out_size; (void)ws_size;

  // Inputs flattened in setup_inputs() insertion order:
  // x, edge_index, then params: convs[0..4] each (w_rel, w_root, b), mu_w, mu_b, ls_w, ls_b
  const float* x    = (const float*)d_in[0];
  const int*   ei   = (const int*)d_in[1];
  const int*   esrc = ei;
  const int*   edst = ei + NEDGES;

  const float *w_rel[5], *w_root[5], *bvec[5];
  for (int l = 0; l < 5; ++l) {
    w_rel[l]  = (const float*)d_in[2 + 3 * l + 0];
    w_root[l] = (const float*)d_in[2 + 3 * l + 1];
    bvec[l]   = (const float*)d_in[2 + 3 * l + 2];
  }
  const float* mu_w = (const float*)d_in[17];
  const float* mu_b = (const float*)d_in[18];
  const float* ls_w = (const float*)d_in[19];
  const float* ls_b = (const float*)d_in[20];

  // Workspace carve (~257 MB): h ping/pong bf16, agg fp32, agg bf16 mirror, bf16 weights
  char* ws = (char*)d_ws;
  auto carve = [&](size_t bytes) -> char* {
    char* p = ws;
    ws += (bytes + 255) & ~(size_t)255;
    return p;
  };
  bf16_t* hA    = (bf16_t*)carve((size_t)NNODES * HDIM * sizeof(bf16_t));
  bf16_t* hB    = (bf16_t*)carve((size_t)NNODES * HDIM * sizeof(bf16_t));
  float*  agg   = (float*) carve((size_t)NNODES * HDIM * sizeof(float));
  bf16_t* aggbf = (bf16_t*)carve((size_t)NNODES * HDIM * sizeof(bf16_t));
  bf16_t* WT[4];
  for (int l = 0; l < 4; ++l) WT[l] = (bf16_t*)carve((size_t)256 * 512 * sizeof(bf16_t));
  bf16_t* WTh   = (bf16_t*)carve((size_t)128 * 256 * sizeof(bf16_t));
  float*  biash = (float*) carve(128 * sizeof(float));

  // ---- weight prep (tiny) ----
  for (int l = 0; l < 4; ++l)
    prep_layer_w<<<(256 * 512 + 255) / 256, 256, 0, stream>>>(w_rel[l + 1], w_root[l + 1], WT[l]);
  prep_heads_w<<<(128 * 256 + 255) / 256, 256, 0, stream>>>(mu_w, ls_w, mu_b, ls_b, WTh, biash);

  // ---- layer 1 (d_in = 5) ----
  fill_zero_f32<<<(NNODES * 5 + 255) / 256, 256, 0, stream>>>(agg, NNODES * 5);
  aggregate_d5<<<(NEDGES + 255) / 256, 256, 0, stream>>>(esrc, edst, x, agg);
  layer1_dense<<<(NNODES * HDIM + 255) / 256, 256, 0, stream>>>(x, agg, w_rel[0], w_root[0],
                                                                bvec[0], hA);

  // ---- layers 2..5: zero agg -> edge aggregation -> bf16 mirror -> fused WMMA GEMM ----
  bf16_t* cur = hA;
  bf16_t* nxt = hB;
  const int n8 = NNODES * HDIM / 8;
  dim3 ggrid((NNODES + 127) / 128, 2);
  for (int l = 0; l < 4; ++l) {
    fill_zero_f32<<<(NNODES * HDIM + 255) / 256, 256, 0, stream>>>(agg, NNODES * HDIM);
    aggregate_h<<<(NEDGES * 16 + 255) / 256, 256, 0, stream>>>(esrc, edst, cur, agg);
    cvt_f32_bf16<<<(n8 + 255) / 256, 256, 0, stream>>>(agg, aggbf, n8);
    gemm_wmma<0><<<ggrid, 256, 0, stream>>>(aggbf, cur, WT[l], bvec[l + 1], nxt, nullptr);
    bf16_t* t = cur; cur = nxt; nxt = t;
  }

  // ---- heads: [N,256] x [256,128] -> (mu | logstd) fp32 ----
  dim3 hgrid((NNODES + 127) / 128, 1);
  gemm_wmma<1><<<hgrid, 256, 0, stream>>>(cur, nullptr, WTh, biash, nullptr, (float*)d_out);
}
